// SpikeDrivenTransformer_83296595738898
// MI455X (gfx1250) — matrix operand order repaired and checked by reading, not compiled, not run on hardware
//
#include <hip/hip_runtime.h>
#include <hip/hip_bf16.h>

typedef __attribute__((ext_vector_type(16))) __bf16 v16bf;
typedef __attribute__((ext_vector_type(8)))  __bf16 v8bf;
typedef __attribute__((ext_vector_type(8)))  float  v8f;

#define T_STEPS 4
#define BATCH   8
#define CDIM    512
#define NDIM    1024
#define BNROWS  (BATCH*NDIM)        // 8192 rows per timestep
#define MROWS   (T_STEPS*BNROWS)    // 32768 total rows
#define CH      2048
#define HEADS   8
#define DHEAD   64
#define BN_EPS  1e-5f

// ---------------------------------------------------------------------------
// f32 -> bf16 weight conversion
// ---------------------------------------------------------------------------
__global__ void k_f32_to_bf16(const float* __restrict__ src,
                              __bf16* __restrict__ dst, int n) {
  int i = blockIdx.x * blockDim.x + threadIdx.x;
  if (i < n) dst[i] = (__bf16)src[i];
}

// ---------------------------------------------------------------------------
// Shortcut LIF + layout transpose: x [T,B,C,N] f32 -> spikes [T*BN, C] bf16
// ---------------------------------------------------------------------------
__global__ __launch_bounds__(256)
void k_lif_transpose(const float* __restrict__ x,
                     unsigned short* __restrict__ spk) {
  __shared__ unsigned short lds[T_STEPS][32][33];
  int bid = blockIdx.x;
  int nb = bid & 31;            // N/32 = 32
  int cb = (bid >> 5) & 15;     // C/32 = 16
  int b  = bid >> 9;            // B = 8
  int c0 = cb * 32, n0 = nb * 32;
  int nl  = threadIdx.x & 31;
  int cl0 = threadIdx.x >> 5;   // 0..7
#pragma unroll
  for (int i = 0; i < 4; ++i) {
    int c = c0 + cl0 + 8 * i;
    float v = 0.f;
#pragma unroll
    for (int t = 0; t < T_STEPS; ++t) {
      float xv = x[((size_t)((t * BATCH + b) * CDIM + c)) * NDIM + n0 + nl];
      v = 0.5f * (v + xv);                       // v += (x - v)/tau, tau=2
      int s = (v >= 0.5f);                       // H(v - v_th)
      lds[t][cl0 + 8 * i][nl] = s ? 0x3F80 : 0;  // bf16 1.0 / 0.0 bits
      v = s ? 0.f : v;                           // hard reset
    }
  }
  __syncthreads();
  int cw  = threadIdx.x & 31;
  int nw0 = threadIdx.x >> 5;
#pragma unroll
  for (int t = 0; t < T_STEPS; ++t)
#pragma unroll
    for (int i = 0; i < 4; ++i) {
      int n = n0 + nw0 + 8 * i;
      spk[((size_t)(t * BNROWS + b * NDIM + n)) * CDIM + c0 + cw] =
          lds[t][cw][nw0 + 8 * i];
    }
}

// ---------------------------------------------------------------------------
// WMMA fragment loaders (per ISA 16-bit A/B layouts, wave32)
// A 16x32: lane<16 -> M=lane, K {0..7, 16..23}; lane>=16 -> K {8..15, 24..31}
// B 32x16: lane<16 -> N=lane, K {0..7, 16..23}; lane>=16 -> K {8..15, 24..31}
// ---------------------------------------------------------------------------
__device__ __forceinline__ v16bf load_a_frag(const __bf16* p) {
  v8bf lo = *(const v8bf*)p;
  v8bf hi = *(const v8bf*)(p + 16);
  return __builtin_shufflevector(lo, hi, 0, 1, 2, 3, 4, 5, 6, 7,
                                 8, 9, 10, 11, 12, 13, 14, 15);
}

__device__ __forceinline__ v8f wmma_bf16(v16bf a, v16bf b, v8f c) {
  return __builtin_amdgcn_wmma_f32_16x16x32_bf16(false, a, false, b,
                                                 (short)0, c, false, false);
}

// Fragment set for one K-chunk of the 16(m) x 32(o) x 4(t) macro-tile.
struct Frags {
  v16bf b0, b1;
  v16bf a[T_STEPS];
};

// ---------------------------------------------------------------------------
// Fused GEMM (bf16 WMMA, f32 acc) + BatchNorm affine + multi-step LIF.
// Macro-tile per wave: 16(m) x 32(o) x 4(t) -> 8 accumulators; the A
// fragment (spikes) is reused across both o sub-tiles.  The K loop is a
// copy-free ping-pong over two fragment sets so loads overlap WMMAs without
// register-rotation movs (which would trigger WMMA->VALU hazard NOPs).
// ---------------------------------------------------------------------------
__global__ __launch_bounds__(128)
void k_gemm_bn_lif(const __bf16* __restrict__ A,
                   const __bf16* __restrict__ W,
                   unsigned short* __restrict__ spk,
                   const float* __restrict__ bnp,   // [4][Cout] g,b,m,v
                   const float* __restrict__ bias,  // may be null
                   int K, int Cout) {
  int lane = threadIdx.x & 31;
  int wid  = blockIdx.x * (blockDim.x >> 5) + (threadIdx.x >> 5);
  int nt_o = Cout >> 5;                   // 32-wide o tiles
  int mtile = wid / nt_o, otile = wid % nt_o;
  int m0 = mtile << 4, o0 = otile << 5;
  int row  = lane & 15;
  int hi16 = lane >> 4;

  const __bf16* bptr0 = W + (size_t)(o0 + row) * K + hi16 * 8;
  const __bf16* bptr1 = bptr0 + (size_t)16 * K;
  const __bf16* ap0 = A + ((size_t)(0 * BNROWS + m0 + row)) * K + hi16 * 8;
  const __bf16* ap1 = A + ((size_t)(1 * BNROWS + m0 + row)) * K + hi16 * 8;
  const __bf16* ap2 = A + ((size_t)(2 * BNROWS + m0 + row)) * K + hi16 * 8;
  const __bf16* ap3 = A + ((size_t)(3 * BNROWS + m0 + row)) * K + hi16 * 8;

  v8f acc[T_STEPS][2] = {};

  auto LOAD = [&](Frags& f, int kk) {
    f.b0 = load_a_frag(bptr0 + kk);
    f.b1 = load_a_frag(bptr1 + kk);
    f.a[0] = load_a_frag(ap0 + kk);
    f.a[1] = load_a_frag(ap1 + kk);
    f.a[2] = load_a_frag(ap2 + kk);
    f.a[3] = load_a_frag(ap3 + kk);
  };
  auto MMA = [&](const Frags& f) {
#pragma unroll
    for (int t = 0; t < T_STEPS; ++t) {
      acc[t][0] = wmma_bf16(f.a[t], f.b0, acc[t][0]);
      acc[t][1] = wmma_bf16(f.a[t], f.b1, acc[t][1]);
    }
  };

  Frags f0, f1;
  LOAD(f0, 0);
  int kk = 0;
  for (; kk < K - 64; kk += 64) {
    LOAD(f1, kk + 32);
    __builtin_prefetch(bptr0 + kk + 96, 0, 3);
    __builtin_prefetch(bptr1 + kk + 96, 0, 3);
    MMA(f0);
    LOAD(f0, kk + 64);
    MMA(f1);
  }
  LOAD(f1, kk + 32);
  MMA(f0);
  MMA(f1);

  int mbase = m0 + (hi16 ? 8 : 0);
#pragma unroll
  for (int j = 0; j < 2; ++j) {
    int o = o0 + 16 * j + row;
    float scale = bnp[o] * rsqrtf(bnp[3 * Cout + o] + BN_EPS);
    float shift = bnp[Cout + o] - bnp[2 * Cout + o] * scale;
    if (bias) shift += bias[o] * scale;
#pragma unroll
    for (int r = 0; r < 8; ++r) {
      int m = mbase + r;
      float v = 0.f;
#pragma unroll
      for (int t = 0; t < T_STEPS; ++t) {
        float y = acc[t][j][r] * scale + shift;
        v = 0.5f * (v + y);
        int s = (v >= 0.5f);
        spk[((size_t)(t * BNROWS + m)) * Cout + o] = s ? 0x3F80 : 0;
        v = s ? 0.f : v;
      }
    }
  }
}

// ---------------------------------------------------------------------------
// proj GEMM: x_att [M,C] x proj_w + bias -> BN -> + identity x[T,B,C,N].
// Same macro-tile / ping-pong; writes x_out f32 + bf16 [M,C].
// ---------------------------------------------------------------------------
__global__ __launch_bounds__(128)
void k_gemm_proj(const __bf16* __restrict__ A,
                 const __bf16* __restrict__ W,
                 const float* __restrict__ bias,
                 const float* __restrict__ bnp,
                 const float* __restrict__ xin,     // identity [T,B,C,N]
                 float* __restrict__ xoutf,         // [M,C] f32
                 __bf16* __restrict__ xoutb) {      // [M,C] bf16
  const int K = CDIM, Cout = CDIM;
  int lane = threadIdx.x & 31;
  int wid  = blockIdx.x * (blockDim.x >> 5) + (threadIdx.x >> 5);
  int nt_o = Cout >> 5;
  int mtile = wid / nt_o, otile = wid % nt_o;
  int m0 = mtile << 4, o0 = otile << 5;
  int row  = lane & 15;
  int hi16 = lane >> 4;

  const __bf16* bptr0 = W + (size_t)(o0 + row) * K + hi16 * 8;
  const __bf16* bptr1 = bptr0 + (size_t)16 * K;
  const __bf16* ap0 = A + ((size_t)(0 * BNROWS + m0 + row)) * K + hi16 * 8;
  const __bf16* ap1 = A + ((size_t)(1 * BNROWS + m0 + row)) * K + hi16 * 8;
  const __bf16* ap2 = A + ((size_t)(2 * BNROWS + m0 + row)) * K + hi16 * 8;
  const __bf16* ap3 = A + ((size_t)(3 * BNROWS + m0 + row)) * K + hi16 * 8;

  v8f acc[T_STEPS][2] = {};

  auto LOAD = [&](Frags& f, int kk) {
    f.b0 = load_a_frag(bptr0 + kk);
    f.b1 = load_a_frag(bptr1 + kk);
    f.a[0] = load_a_frag(ap0 + kk);
    f.a[1] = load_a_frag(ap1 + kk);
    f.a[2] = load_a_frag(ap2 + kk);
    f.a[3] = load_a_frag(ap3 + kk);
  };
  auto MMA = [&](const Frags& f) {
#pragma unroll
    for (int t = 0; t < T_STEPS; ++t) {
      acc[t][0] = wmma_bf16(f.a[t], f.b0, acc[t][0]);
      acc[t][1] = wmma_bf16(f.a[t], f.b1, acc[t][1]);
    }
  };

  Frags f0, f1;
  LOAD(f0, 0);
  int kk = 0;
  for (; kk < K - 64; kk += 64) {
    LOAD(f1, kk + 32);
    __builtin_prefetch(bptr0 + kk + 96, 0, 3);
    __builtin_prefetch(bptr1 + kk + 96, 0, 3);
    MMA(f0);
    LOAD(f0, kk + 64);
    MMA(f1);
  }
  LOAD(f1, kk + 32);
  MMA(f0);
  MMA(f1);

  int mbase = m0 + (hi16 ? 8 : 0);
#pragma unroll
  for (int j = 0; j < 2; ++j) {
    int o = o0 + 16 * j + row;
    float scale = bnp[o] * rsqrtf(bnp[3 * Cout + o] + BN_EPS);
    float shift = bnp[Cout + o] - bnp[2 * Cout + o] * scale + bias[o] * scale;
#pragma unroll
    for (int r = 0; r < 8; ++r) {
      int m = mbase + r;
      int b = m >> 10;            // N = 1024
      int n = m & (NDIM - 1);
#pragma unroll
      for (int t = 0; t < T_STEPS; ++t) {
        float id = xin[((size_t)((t * BATCH + b) * CDIM + o)) * NDIM + n];
        float val = acc[t][j][r] * scale + shift + id;
        size_t idx = ((size_t)(t * BNROWS + m)) * CDIM + o;
        xoutf[idx] = val;
        xoutb[idx] = (__bf16)val;
      }
    }
  }
}

// ---------------------------------------------------------------------------
// kv[t,b,c] = sum_n k_spk * v_spk   (lanes stride c -> coalesced loads)
// ---------------------------------------------------------------------------
__global__ void k_kv_reduce(const __bf16* __restrict__ ks,
                            const __bf16* __restrict__ vs,
                            float* __restrict__ kv) {
  int idx = blockIdx.x * blockDim.x + threadIdx.x;  // T*B*C = 16384
  if (idx >= T_STEPS * BATCH * CDIM) return;
  int c = idx & (CDIM - 1);
  int b = (idx >> 9) & (BATCH - 1);
  int t = idx >> 12;
  size_t base = ((size_t)(t * BNROWS + b * NDIM)) * CDIM + c;
  float s = 0.f;
  for (int n = 0; n < NDIM; ++n) {
    s += (float)ks[base + (size_t)n * CDIM] * (float)vs[base + (size_t)n * CDIM];
  }
  kv[idx] = s;
}

// ---------------------------------------------------------------------------
// talking heads (8x8 mix) + LIF over T -> kv spikes [T,B,C] f32 (0/1)
// ---------------------------------------------------------------------------
__global__ void k_talking_lif(const float* __restrict__ kv,
                              const float* __restrict__ thw,
                              float* __restrict__ kvs) {
  int idx = blockIdx.x * blockDim.x + threadIdx.x;  // B*C = 4096
  if (idx >= BATCH * CDIM) return;
  int c = idx & (CDIM - 1);
  int b = idx >> 9;
  int g = c >> 6, dd = c & (DHEAD - 1);
  float v = 0.f;
#pragma unroll
  for (int t = 0; t < T_STEPS; ++t) {
    float y = 0.f;
#pragma unroll
    for (int h = 0; h < HEADS; ++h)
      y += thw[g * HEADS + h] *
           kv[(size_t)t * BATCH * CDIM + b * CDIM + h * DHEAD + dd];
    v = 0.5f * (v + y);
    int s = (v >= 0.5f);
    kvs[(size_t)t * BATCH * CDIM + b * CDIM + c] = s ? 1.f : 0.f;
    v = s ? 0.f : v;
  }
}

// ---------------------------------------------------------------------------
// x_att = q_spk * kv_spk (broadcast over n); product of spikes is binary.
// ---------------------------------------------------------------------------
__global__ void k_xatt(const __bf16* __restrict__ q,
                       const float* __restrict__ kvs,
                       __bf16* __restrict__ xa) {
  size_t idx = (size_t)blockIdx.x * blockDim.x + threadIdx.x;  // M*C
  int c = (int)(idx & (CDIM - 1));
  size_t m = idx >> 9;
  int t = (int)(m >> 13);              // BN = 8192
  int b = (int)((m >> 10) & 7);
  float prod = (float)q[idx] * kvs[(size_t)(t * BATCH + b) * CDIM + c];
  xa[idx] = (__bf16)prod;
}

// ---------------------------------------------------------------------------
// Final: out[t,b,c,n] = x_out[m,c] + fc2_spike[m,c] via LDS tile transpose.
// ---------------------------------------------------------------------------
__global__ __launch_bounds__(256)
void k_final(const float* __restrict__ xoutf,
             const __bf16* __restrict__ spk,
             float* __restrict__ out) {
  __shared__ float lds[32][33];
  int bid = blockIdx.x;
  int nb = bid & 31;
  int cb = (bid >> 5) & 15;
  int b  = (bid >> 9) & 7;
  int t  = bid >> 12;
  int c0 = cb * 32, n0 = nb * 32;
  int cl  = threadIdx.x & 31;
  int nl0 = threadIdx.x >> 5;
#pragma unroll
  for (int i = 0; i < 4; ++i) {
    int n = n0 + nl0 + 8 * i;
    size_t idx = ((size_t)(t * BNROWS + b * NDIM + n)) * CDIM + c0 + cl;
    lds[nl0 + 8 * i][cl] = xoutf[idx] + (float)spk[idx];
  }
  __syncthreads();
  int nw  = threadIdx.x & 31;
  int cw0 = threadIdx.x >> 5;
#pragma unroll
  for (int i = 0; i < 4; ++i) {
    int c = c0 + cw0 + 8 * i;
    out[((size_t)((t * BATCH + b) * CDIM + c)) * NDIM + n0 + nw] =
        lds[nw][cw0 + 8 * i];
  }
}

// ---------------------------------------------------------------------------
extern "C" void kernel_launch(void* const* d_in, const int* in_sizes, int n_in,
                              void* d_out, int out_size, void* d_ws,
                              size_t ws_size, hipStream_t stream) {
  const float* x     = (const float*)d_in[0];
  const float* q_w   = (const float*)d_in[1];
  const float* q_bn  = (const float*)d_in[2];
  const float* k_w   = (const float*)d_in[3];
  const float* k_bn  = (const float*)d_in[4];
  const float* v_w   = (const float*)d_in[5];
  const float* v_bn  = (const float*)d_in[6];
  const float* th_w  = (const float*)d_in[7];
  const float* p_w   = (const float*)d_in[8];
  const float* p_b   = (const float*)d_in[9];
  const float* p_bn  = (const float*)d_in[10];
  const float* f1_w  = (const float*)d_in[11];
  const float* f1_b  = (const float*)d_in[12];
  const float* f1_bn = (const float*)d_in[13];
  const float* f2_w  = (const float*)d_in[14];
  const float* f2_b  = (const float*)d_in[15];
  const float* f2_bn = (const float*)d_in[16];

  char* ws = (char*)d_ws;
  const size_t SPK_BYTES = (size_t)MROWS * CDIM * 2;  // 32 MB
  const size_t OFF_WQ    = 0;                          // 512 KB each conv wgt
  const size_t OFF_WK    = OFF_WQ + (size_t)CDIM * CDIM * 2;
  const size_t OFF_WV    = OFF_WK + (size_t)CDIM * CDIM * 2;
  const size_t OFF_WP    = OFF_WV + (size_t)CDIM * CDIM * 2;
  const size_t OFF_WF1   = OFF_WP + (size_t)CDIM * CDIM * 2;
  const size_t OFF_WF2   = OFF_WF1 + (size_t)CH * CDIM * 2;
  const size_t OFF_KV    = OFF_WF2 + (size_t)CH * CDIM * 2;
  const size_t OFF_KVS   = OFF_KV + (size_t)T_STEPS * BATCH * CDIM * 4;
  const size_t OFF_XS    = 8ull << 20;
  const size_t OFF_Q     = OFF_XS + SPK_BYTES;
  const size_t OFF_K     = OFF_Q + SPK_BYTES;
  const size_t OFF_V     = OFF_K + SPK_BYTES;
  const size_t OFF_XATT  = OFF_XS;                     // reuse (xs dead)
  const size_t OFF_XOUTF = OFF_V + SPK_BYTES;          // [M,C] f32, 64 MB
  const size_t OFF_XOUTB = OFF_K;                      // reuse k_spk
  const size_t OFF_HDN   = OFF_XOUTF + (size_t)MROWS * CDIM * 4;  // 128 MB
  const size_t OFF_O2    = OFF_V;                      // reuse v_spk

  const int CC = CDIM * CDIM, CHC = CH * CDIM;
  k_f32_to_bf16<<<(CC + 255) / 256, 256, 0, stream>>>(q_w, (__bf16*)(ws + OFF_WQ), CC);
  k_f32_to_bf16<<<(CC + 255) / 256, 256, 0, stream>>>(k_w, (__bf16*)(ws + OFF_WK), CC);
  k_f32_to_bf16<<<(CC + 255) / 256, 256, 0, stream>>>(v_w, (__bf16*)(ws + OFF_WV), CC);
  k_f32_to_bf16<<<(CC + 255) / 256, 256, 0, stream>>>(p_w, (__bf16*)(ws + OFF_WP), CC);
  k_f32_to_bf16<<<(CHC + 255) / 256, 256, 0, stream>>>(f1_w, (__bf16*)(ws + OFF_WF1), CHC);
  k_f32_to_bf16<<<(CHC + 255) / 256, 256, 0, stream>>>(f2_w, (__bf16*)(ws + OFF_WF2), CHC);

  // shortcut LIF + transpose to [M, C] bf16 spikes
  k_lif_transpose<<<BATCH * (CDIM / 32) * (NDIM / 32), 256, 0, stream>>>(
      x, (unsigned short*)(ws + OFF_XS));

  // q/k/v: GEMM + BN + LIF  (waves = 512 m-tiles * 16 o-tiles = 8192)
  k_gemm_bn_lif<<<2048, 128, 0, stream>>>((__bf16*)(ws + OFF_XS),
      (__bf16*)(ws + OFF_WQ), (unsigned short*)(ws + OFF_Q), q_bn, nullptr, CDIM, CDIM);
  k_gemm_bn_lif<<<2048, 128, 0, stream>>>((__bf16*)(ws + OFF_XS),
      (__bf16*)(ws + OFF_WK), (unsigned short*)(ws + OFF_K), k_bn, nullptr, CDIM, CDIM);
  k_gemm_bn_lif<<<2048, 128, 0, stream>>>((__bf16*)(ws + OFF_XS),
      (__bf16*)(ws + OFF_WV), (unsigned short*)(ws + OFF_V), v_bn, nullptr, CDIM, CDIM);

  // attention core
  k_kv_reduce<<<64, 256, 0, stream>>>((const __bf16*)(ws + OFF_K),
      (const __bf16*)(ws + OFF_V), (float*)(ws + OFF_KV));
  k_talking_lif<<<16, 256, 0, stream>>>((const float*)(ws + OFF_KV), th_w,
      (float*)(ws + OFF_KVS));
  k_xatt<<<(int)(((size_t)MROWS * CDIM) / 256), 256, 0, stream>>>(
      (const __bf16*)(ws + OFF_Q), (const float*)(ws + OFF_KVS),
      (__bf16*)(ws + OFF_XATT));

  // proj GEMM + bias + BN + residual
  k_gemm_proj<<<2048, 128, 0, stream>>>((const __bf16*)(ws + OFF_XATT),
      (const __bf16*)(ws + OFF_WP), p_b, p_bn, x,
      (float*)(ws + OFF_XOUTF), (__bf16*)(ws + OFF_XOUTB));

  // MLP: fc1 (512 * 64 o-tiles = 32768 waves) and fc2 (K=2048)
  k_gemm_bn_lif<<<8192, 128, 0, stream>>>((__bf16*)(ws + OFF_XOUTB),
      (__bf16*)(ws + OFF_WF1), (unsigned short*)(ws + OFF_HDN), f1_bn, f1_b, CDIM, CH);
  k_gemm_bn_lif<<<2048, 128, 0, stream>>>((__bf16*)(ws + OFF_HDN),
      (__bf16*)(ws + OFF_WF2), (unsigned short*)(ws + OFF_O2), f2_bn, f2_b, CH, CDIM);

  // final: spike + residual, back to [T,B,C,H,W]
  k_final<<<T_STEPS * BATCH * (CDIM / 32) * (NDIM / 32), 256, 0, stream>>>(
      (const float*)(ws + OFF_XOUTF), (const __bf16*)(ws + OFF_O2),
      (float*)d_out);
}
